// ConvRNN_3324304687352
// MI455X (gfx1250) — matrix operand order repaired and checked by reading, not compiled, not run on hardware
//
#include <hip/hip_runtime.h>
#include <hip/hip_bf16.h>
#include <math.h>

// ---------------------------------------------------------------------------
// ConvGRU for MI455X (gfx1250):
//  - implicit-GEMM convs via v_wmma_f32_16x16x32_f16 (f32 accumulate)
//  - activation tiles staged into LDS by the Tensor Data Mover
//    (tensor_load_to_lds + s_wait_tensorcnt), NHWC f16 activation mirrors
// ---------------------------------------------------------------------------

typedef __attribute__((ext_vector_type(16))) _Float16     v16h;
typedef __attribute__((ext_vector_type(8)))  _Float16     v8h;
typedef __attribute__((ext_vector_type(8)))  float        v8f;
typedef __attribute__((ext_vector_type(4)))  unsigned int v4u;
typedef __attribute__((ext_vector_type(8)))  int          v8i;
typedef __attribute__((ext_vector_type(4)))  int          v4i;

#if defined(__HIP_DEVICE_COMPILE__) && __has_builtin(__builtin_amdgcn_tensor_load_to_lds)
#define USE_TDM 1
#else
#define USE_TDM 0
#endif

__device__ __forceinline__ float sigmoidf_(float x) { return 1.0f / (1.0f + expf(-x)); }
__device__ __forceinline__ float relu6f_(float x)   { return fminf(fmaxf(x, 0.0f), 6.0f); }

// ---------------------------------------------------------------------------
// Weight packing: w[CO][CIN][KS][KS] f32 -> f16 WMMA A-fragments.
//   lane L (half h = L>>4, ln = L&15): element e -> M = mt*16+ln,
//   K-in-tile kk = h*8 + (e<8 ? e : e+8)   (V0-3: K=h*8+0..7, V4-7: 16+h*8+0..7)
// kt enumerates (tap, cin_block32); CIN % 32 == 0 so tiles never straddle taps.
// ---------------------------------------------------------------------------
__global__ void prep_w(const float* __restrict__ w, _Float16* __restrict__ apk,
                       int CO, int CIN, int KS) {
    int idx = blockIdx.x * 256 + threadIdx.x;
    int KT = KS * KS * (CIN / 32);
    int total = (CO / 16) * KT * 512;
    if (idx >= total) return;
    int e    = idx & 15;
    int lane = (idx >> 4) & 31;
    int frag = idx >> 9;
    int kt = frag % KT;
    int mt = frag / KT;
    int h  = lane >> 4;
    int kk = h * 8 + (e < 8 ? e : e + 8);
    int tap = kt / (CIN / 32);
    int cin = (kt % (CIN / 32)) * 32 + kk;
    int ky = tap / KS, kx = tap % KS;
    int co = mt * 16 + (lane & 15);
    apk[idx] = (_Float16)w[(((size_t)co * CIN + cin) * KS + ky) * KS + kx];
}

// NCHW f32 -> NHWC f16 (one-time conversion of the input sequence x).
__global__ void to_nhwc_f16(const float* __restrict__ src, _Float16* __restrict__ dst,
                            int C, size_t total /* B*C*4096 */) {
    size_t i = (size_t)blockIdx.x * 256 + threadIdx.x;
    if (i >= total) return;
    int x  = (int)(i & 63);
    int yy = (int)((i >> 6) & 63);
    int c  = (int)((i >> 12) % C);
    size_t b = i / ((size_t)C * 4096);
    dst[(((b * 64 + yy) * 64 + x) * C) + c] = (_Float16)src[i];
}

// ---------------------------------------------------------------------------
// Implicit-GEMM conv, stride 1, 64x64 "same" padding. Input is NHWC f16.
// Block = 128 threads (4 wave32s). blockIdx = (row y, image b, 64-chan chunk).
// Patch rows are DMA'd into LDS by the TDM (pad_enable inserts the +8-half
// channel padding; image-edge rows/cols stay at the pre-zeroed value).
// ---------------------------------------------------------------------------
template <int KS, int CIN>
__global__ __launch_bounds__(128)
void conv_wmma(const _Float16* __restrict__ in,     // [B, 64, 64, CIN] NHWC f16
               const _Float16* __restrict__ apk,    // packed A fragments
               float* __restrict__ out,             // [B, CO, 64, 64] raw conv
               float* __restrict__ stats,           // [2*CO] sum, sumsq
               int CO, int B) {
    constexpr int PAD    = KS / 2;
    constexpr int PW     = 64 + 2 * PAD;
    constexpr int CINP   = CIN + 8;                 // +16B pad per column slot
    constexpr int KT     = KS * KS * (CIN / 32);
    constexpr int MT     = 4;
    constexpr int PADIV  = (CIN == 128) ? 5 : 4;    // pad interval: CIN*2B in 2^(n+1) DW
    constexpr int PADAMT = 3;                       // pad amount: 4 DW = 16B = 8 halves

    __shared__ __align__(16) _Float16 lds[KS * PW * CINP];

    const int tid = threadIdx.x;
    const int y   = blockIdx.x;
    const int b   = blockIdx.y;
    const int cz  = blockIdx.z;

    // Pre-zero the patch (covers image-edge padding); 16B LDS stores.
    {
        v8h zz;
#pragma unroll
        for (int i = 0; i < 8; ++i) zz[i] = (_Float16)0.0f;
        for (int i = tid * 8; i < KS * PW * CINP; i += 128 * 8)
            *(v8h*)&lds[i] = zz;
    }
    __syncthreads();

#if USE_TDM
    if (tid < 32) {  // TDM ignores EXEC; issue from one wave only
        unsigned lds_base = (unsigned)(unsigned long long)(&lds[0]);
#pragma unroll
        for (int r = 0; r < KS; ++r) {
            int iy = y + r - PAD;
            if (iy >= 0 && iy < 64) {
                unsigned long long ga = (unsigned long long)in +
                    (((unsigned long long)b * 64 + (unsigned)iy) * 64) *
                        (unsigned long long)(CIN * 2);
                // D# group 0: count=1 | lds_addr | global_addr | type=2
                v4u g0;
                g0[0] = 1u;
                g0[1] = lds_base + (unsigned)((r * PW + PAD) * CINP * 2);
                g0[2] = (unsigned)(ga & 0xffffffffull);
                g0[3] = (unsigned)((ga >> 32) & 0x01ffffffull) | (2u << 30);
                // D# group 1: data_size=2B, pad_enable, 2D tile CIN x 64
                v8i g1;
                g1[0] = (int)((1u << 16) | (1u << 20) |
                              ((unsigned)PADIV << 22) | ((unsigned)PADAMT << 25));
                g1[1] = (int)((unsigned)CIN << 16);  // tensor_dim0[15:0]
                g1[2] = (int)(64u << 16);            // tensor_dim1[15:0]
                g1[3] = (int)((unsigned)CIN << 16);  // tile_dim0
                g1[4] = (int)64;                     // tile_dim1 (tile_dim2=0)
                g1[5] = (int)CIN;                    // tensor_dim0_stride[31:0]
                g1[6] = 0;
                g1[7] = 0;
                v4i g2; g2[0] = 0; g2[1] = 0; g2[2] = 0; g2[3] = 0;
                v4i g3; g3[0] = 0; g3[1] = 0; g3[2] = 0; g3[3] = 0;
#if __clang_major__ >= 23
                v8i g4;
#pragma unroll
                for (int q = 0; q < 8; ++q) g4[q] = 0;
                __builtin_amdgcn_tensor_load_to_lds(g0, g1, g2, g3, g4, 0);
#else
                __builtin_amdgcn_tensor_load_to_lds(g0, g1, g2, g3, 0);
#endif
            }
        }
#if __has_builtin(__builtin_amdgcn_s_wait_tensorcnt)
        __builtin_amdgcn_s_wait_tensorcnt(0);
#else
        asm volatile("s_wait_tensorcnt 0x0" ::: "memory");
#endif
    }
#else
    // Fallback: cooperative f16 copy (NHWC => channel-fastest is coalesced).
    for (int i = tid; i < KS * PW * CIN; i += 128) {
        int ci = i % CIN;
        int c  = (i / CIN) % PW;
        int r  = i / (CIN * PW);
        int iy = y + r - PAD;
        int ix = c - PAD;
        if (iy >= 0 && iy < 64 && ix >= 0 && ix < 64)
            lds[(r * PW + c) * CINP + ci] =
                in[((((size_t)b * 64 + iy) * 64 + ix) * CIN) + ci];
    }
#endif
    __syncthreads();

    const int lane = tid & 31;
    const int wv   = tid >> 5;
    const int x0   = wv * 16;
    const int hf   = lane >> 4;
    const int ln   = lane & 15;

    v8f acc[MT];
#pragma unroll
    for (int m = 0; m < MT; ++m)
#pragma unroll
        for (int r = 0; r < 8; ++r) acc[m][r] = 0.0f;

#pragma unroll 2
    for (int kt = 0; kt < KT; ++kt) {
        int tap = kt / (CIN / 32);
        int cb  = (kt % (CIN / 32)) * 32;
        int ky  = tap / KS, kx = tap % KS;
        // B fragment: lane-half h holds K = h*8 + e (e<8) and 16 + h*8 + (e-8)
        // (e>=8) in the channel-contiguous LDS layout: two 16B ds loads.
        const _Float16* bp = &lds[((ky * PW) + x0 + ln + kx) * CINP + cb + hf * 8];
        v8h blo = *(const v8h*)bp;
        v8h bhi = *(const v8h*)(bp + 16);
        v16h bfrag;
#pragma unroll
        for (int i = 0; i < 8; ++i) { bfrag[i] = blo[i]; bfrag[i + 8] = bhi[i]; }

#pragma unroll
        for (int m = 0; m < MT; ++m) {
            int mt = cz * MT + m;
            v16h afrag = *(const v16h*)&apk[((size_t)(mt * KT + kt)) * 512 + (size_t)lane * 16];
            acc[m] = __builtin_amdgcn_wmma_f32_16x16x32_f16(
                false, afrag, false, bfrag, (short)0, acc[m], false, false);
        }
    }

    // C/D layout: lanes 0-15: VGPR r -> M=r ; lanes 16-31: M=8+r ; N = ln.
    size_t obase = (((size_t)b * CO) * 64 + y) * 64;
#pragma unroll
    for (int m = 0; m < MT; ++m) {
        int co0 = (cz * MT + m) * 16 + hf * 8;
#pragma unroll
        for (int r = 0; r < 8; ++r) {
            float v = acc[m][r];
            out[obase + (size_t)(co0 + r) * 4096 + x0 + ln] = v;
            float s = v, s2 = v * v;
#pragma unroll
            for (int d = 1; d < 16; d <<= 1) {
                s  += __shfl_xor(s, d);
                s2 += __shfl_xor(s2, d);
            }
            if (ln == 0) {
                atomicAdd(&stats[co0 + r], s);
                atomicAdd(&stats[CO + co0 + r], s2);
            }
        }
    }
}

// ---------------------------------------------------------------------------
__global__ void zero_f32(float* p, int n) {
    int i = blockIdx.x * 256 + threadIdx.x;
    if (i < n) p[i] = 0.0f;
}

// In place: stats[sum, sumsq] -> [scale, shift]. (Conv bias is a per-channel
// constant, cancelled exactly by BN mean subtraction, so it is omitted.)
__global__ void bn_finalize(float* stats, const float* __restrict__ gamma,
                            const float* __restrict__ beta, int CO, float inv_count) {
    int c = blockIdx.x * 64 + threadIdx.x;
    if (c >= CO) return;
    float m  = stats[c] * inv_count;
    float vv = stats[CO + c] * inv_count - m * m;
    float sc = gamma[c] * rsqrtf(vv + 1e-5f);
    stats[c]      = sc;
    stats[CO + c] = beta[c] - m * sc;
}

__global__ void bn_apply_inplace(float* y, const float* __restrict__ ss, int CO, size_t total) {
    size_t i = (size_t)blockIdx.x * 256 + threadIdx.x;
    if (i >= total) return;
    int c = (int)((i >> 12) % CO);
    y[i] = y[i] * ss[c] + ss[CO + c];
}

// h0 = sigmoid(y0[:, :128]) * tanh(y0[:, 256:384]) ; also emit NHWC f16 mirror
__global__ void h0_kernel(const float* __restrict__ y, float* __restrict__ hout,
                          _Float16* __restrict__ h16) {
    int i = blockIdx.x * 256 + threadIdx.x;
    if (i >= 4 * 128 * 4096) return;
    int p = i & 4095, c = (i >> 12) & 127, n = i >> 19;
    float xz = y[((size_t)n * 384 + c) * 4096 + p];
    float xh = y[((size_t)n * 384 + 256 + c) * 4096 + p];
    float h  = sigmoidf_(xz) * tanhf(xh);
    hout[i] = h;
    h16[((size_t)n * 4096 + p) * 128 + c] = (_Float16)h;
}

// tmp = relu6(BN(conv_zr)) + x_zr ; z = sig(tmp[:, :128]) ; r = sig(tmp[:, 128:])
// rh = r * h_{t-1}  (NHWC f16, feeds conv_hh)
__global__ void zr_apply(const float* __restrict__ zr_raw, const float* __restrict__ ss,
                         const float* __restrict__ yt, const float* __restrict__ hprev,
                         float* __restrict__ zbuf, _Float16* __restrict__ rh16) {
    int i = blockIdx.x * 256 + threadIdx.x;
    if (i >= 4 * 128 * 4096) return;
    int p = i & 4095, c = (i >> 12) & 127, n = i >> 19;
    size_t base = ((size_t)n * 256 + c) * 4096 + p;
    float rz = zr_raw[base];
    float rr = zr_raw[base + (size_t)128 * 4096];
    float tz = relu6f_(rz * ss[c] + ss[256 + c])             + yt[((size_t)n * 384 + c) * 4096 + p];
    float tr = relu6f_(rr * ss[128 + c] + ss[256 + 128 + c]) + yt[((size_t)n * 384 + 128 + c) * 4096 + p];
    zbuf[i] = sigmoidf_(tz);
    rh16[((size_t)n * 4096 + p) * 128 + c] = (_Float16)(sigmoidf_(tr) * hprev[i]);
}

// hh = relu6(BN(conv_hh)) + x_h ; h_t = (1-z) h_{t-1} + z tanh(hh)
__global__ void hh_apply(const float* __restrict__ hh_raw, const float* __restrict__ ss,
                         const float* __restrict__ yt, const float* __restrict__ hprev,
                         const float* __restrict__ zbuf, float* __restrict__ hout,
                         _Float16* __restrict__ h16) {
    int i = blockIdx.x * 256 + threadIdx.x;
    if (i >= 4 * 128 * 4096) return;
    int p = i & 4095, c = (i >> 12) & 127, n = i >> 19;
    float hv = relu6f_(hh_raw[i] * ss[c] + ss[128 + c]) + yt[((size_t)n * 384 + 256 + c) * 4096 + p];
    float z  = zbuf[i];
    float h  = (1.0f - z) * hprev[i] + z * tanhf(hv);
    hout[i] = h;
    h16[((size_t)n * 4096 + p) * 128 + c] = (_Float16)h;
}

// ---------------------------------------------------------------------------
extern "C" void kernel_launch(void* const* d_in, const int* in_sizes, int n_in,
                              void* d_out, int out_size, void* d_ws, size_t ws_size,
                              hipStream_t stream) {
    (void)in_sizes; (void)n_in; (void)out_size; (void)ws_size;
    const float* x      = (const float*)d_in[0];
    const float* w_x2h  = (const float*)d_in[1];
    const float* g_x2h  = (const float*)d_in[3];
    const float* bt_x2h = (const float*)d_in[4];
    const float* w_zr   = (const float*)d_in[5];
    const float* g_zr   = (const float*)d_in[7];
    const float* bt_zr  = (const float*)d_in[8];
    const float* w_hh   = (const float*)d_in[9];
    const float* g_hh   = (const float*)d_in[11];
    const float* bt_hh  = (const float*)d_in[12];
    float* out = (float*)d_out;

    char* wsp = (char*)d_ws;
    auto carve = [&](size_t bytes) -> char* {
        char* p = wsp;
        wsp += (bytes + 255) & ~(size_t)255;
        return p;
    };
    float*    y      = (float*)carve((size_t)32 * 384 * 4096 * 4);  // x2h raw -> BN'd
    float*    zr_raw = (float*)carve((size_t)4 * 256 * 4096 * 4);
    float*    hh_raw = (float*)carve((size_t)4 * 128 * 4096 * 4);
    float*    zbuf   = (float*)carve((size_t)4 * 128 * 4096 * 4);
    float*    stats  = (float*)carve(2 * 384 * 4);
    _Float16* xh16   = (_Float16*)carve((size_t)32 * 4096 * 64 * 2);  // x NHWC f16
    _Float16* h16    = (_Float16*)carve((size_t)4 * 4096 * 128 * 2);  // h_{t-1} NHWC f16
    _Float16* rh16   = (_Float16*)carve((size_t)4 * 4096 * 128 * 2);  // r*h NHWC f16
    _Float16* apx    = (_Float16*)carve((size_t)24 * 50 * 512 * 2);
    _Float16* apz    = (_Float16*)carve((size_t)16 * 36 * 512 * 2);
    _Float16* aph    = (_Float16*)carve((size_t)8 * 36 * 512 * 2);

    // Pack weights into WMMA A-fragment layout; convert x to NHWC f16.
    { int tot = 24 * 50 * 512; prep_w<<<(tot + 255) / 256, 256, 0, stream>>>(w_x2h, apx, 384, 64, 5); }
    { int tot = 16 * 36 * 512; prep_w<<<(tot + 255) / 256, 256, 0, stream>>>(w_zr,  apz, 256, 128, 3); }
    { int tot =  8 * 36 * 512; prep_w<<<(tot + 255) / 256, 256, 0, stream>>>(w_hh,  aph, 128, 128, 3); }
    {
        size_t tot = (size_t)32 * 64 * 4096;
        to_nhwc_f16<<<(unsigned)((tot + 255) / 256), 256, 0, stream>>>(x, xh16, 64, tot);
    }

    // --- x2h: 5x5 conv over all 32 images, then BN ---
    zero_f32<<<3, 256, 0, stream>>>(stats, 768);
    conv_wmma<5, 64><<<dim3(64, 32, 6), 128, 0, stream>>>(xh16, apx, y, stats, 384, 32);
    bn_finalize<<<6, 64, 0, stream>>>(stats, g_x2h, bt_x2h, 384, 1.0f / (32.0f * 4096.0f));
    {
        size_t tot = (size_t)32 * 384 * 4096;
        bn_apply_inplace<<<(unsigned)((tot + 255) / 256), 256, 0, stream>>>(y, stats, 384, tot);
    }

    const int HS = 4 * 128 * 4096;  // one h-state slab (one time slice of d_out)
    h0_kernel<<<(HS + 255) / 256, 256, 0, stream>>>(y, out, h16);

    for (int t = 1; t < 8; ++t) {
        const float* yt    = y + (size_t)t * 4 * 384 * 4096;
        const float* hprev = out + (size_t)(t - 1) * HS;
        float*       hcur  = out + (size_t)t * HS;

        zero_f32<<<3, 256, 0, stream>>>(stats, 768);
        conv_wmma<3, 128><<<dim3(64, 4, 4), 128, 0, stream>>>(h16, apz, zr_raw, stats, 256, 4);
        bn_finalize<<<4, 64, 0, stream>>>(stats, g_zr, bt_zr, 256, 1.0f / (4.0f * 4096.0f));
        zr_apply<<<(HS + 255) / 256, 256, 0, stream>>>(zr_raw, stats, yt, hprev, zbuf, rh16);

        zero_f32<<<3, 256, 0, stream>>>(stats, 768);
        conv_wmma<3, 128><<<dim3(64, 4, 2), 128, 0, stream>>>(rh16, aph, hh_raw, stats, 128, 4);
        bn_finalize<<<2, 64, 0, stream>>>(stats, g_hh, bt_hh, 128, 1.0f / (4.0f * 4096.0f));
        hh_apply<<<(HS + 255) / 256, 256, 0, stream>>>(hh_raw, stats, yt, hprev, zbuf, hcur, h16);
    }
}
